// Attention_26912265076816
// MI455X (gfx1250) — matrix operand order repaired and checked by reading, not compiled, not run on hardware
//
#include <hip/hip_runtime.h>
#include <hip/hip_bf16.h>
#include <math.h>

typedef __attribute__((ext_vector_type(16))) _Float16 v16h;
typedef __attribute__((ext_vector_type(8)))  float    v8f;

namespace {
constexpr int B_      = 2;
constexpr int DIM_    = 4096;
constexpr int H_      = 32;
constexpr int HKV_    = 8;
constexpr int D_      = 128;
constexpr int R_      = 4;          // H / HKV
constexpr int PAGE_   = 16;
constexpr int START_  = 32768;
constexpr int WINDOW_ = 4096;
constexpr int PREFIX_ = START_ - WINDOW_;   // 28672
constexpr int NPAGES_ = PREFIX_ / PAGE_;    // 1792
constexpr int TOPT_   = 256;                // top pages kept per (b,hkv,r)
constexpr int NCHUNK_ = 17;                 // 8 prefix + 8 suffix + 1 new token
constexpr float SCALE_ = 0.08838834764831845f; // 1/sqrt(128)

// workspace layout (in floats)
constexpr size_t OFF_Q   = 0;                                  // B*H*D      = 8192
constexpr size_t OFF_KN  = OFF_Q  + (size_t)B_ * H_ * D_;      // B*HKV*D    = 2048
constexpr size_t OFF_VN  = OFF_KN + (size_t)B_ * HKV_ * D_;
constexpr size_t OFF_SC  = OFF_VN + (size_t)B_ * HKV_ * D_;    // B*HKV*R*NPAGES = 114688
constexpr size_t OFF_AO  = OFF_SC + (size_t)B_ * HKV_ * R_ * NPAGES_; // B*H*D
constexpr size_t OFF_PO  = OFF_AO + (size_t)B_ * H_ * D_;      // B*H*NCHUNK*D
constexpr size_t OFF_PL  = OFF_PO + (size_t)B_ * H_ * NCHUNK_ * D_;   // B*H*NCHUNK
constexpr size_t OFF_TOP = OFF_PL + (size_t)B_ * H_ * NCHUNK_; // ints: B*HKV*R*TOPT
} // namespace

// ---------------------------------------------------------------------------
// GEMV: y[b][row] = dot(x[b][:], W[row][:]) ; W is (rows, DIM) row-major.
// One 256-thread block per output element; coalesced weight reads.
// ---------------------------------------------------------------------------
__global__ void gemv_kernel(const float* __restrict__ x,
                            const float* __restrict__ W,
                            float* __restrict__ y, int rows) {
  const int row = blockIdx.x % rows;
  const int b   = blockIdx.x / rows;
  const float* xr = x + (size_t)b * DIM_;
  const float* wr = W + (size_t)row * DIM_;
  float p = 0.f;
  for (int i = threadIdx.x; i < DIM_; i += 256) p += xr[i] * wr[i];
  __shared__ float red[256];
  red[threadIdx.x] = p;
  __syncthreads();
  for (int s = 128; s > 0; s >>= 1) {
    if (threadIdx.x < s) red[threadIdx.x] += red[threadIdx.x + s];
    __syncthreads();
  }
  if (threadIdx.x == 0) y[(size_t)b * rows + row] = red[0];
}

// ---------------------------------------------------------------------------
// RoPE on q (B*H heads) and the new k (B*HKV heads), pairwise (even, odd).
// ---------------------------------------------------------------------------
__global__ void rope_kernel(float* __restrict__ q, float* __restrict__ kn,
                            const float* __restrict__ cosv,
                            const float* __restrict__ sinv) {
  const int i  = blockIdx.x * 256 + threadIdx.x;
  const int NQ = B_ * H_ * (D_ / 2);
  const int NK = B_ * HKV_ * (D_ / 2);
  if (i < NQ + NK) {
    float* t;
    int d2;
    if (i < NQ) { d2 = i % (D_ / 2); t = q  + (size_t)(i  / (D_ / 2)) * D_ + 2 * d2; }
    else { int j = i - NQ; d2 = j % (D_ / 2); t = kn + (size_t)(j / (D_ / 2)) * D_ + 2 * d2; }
    const float c = cosv[d2], s = sinv[d2];
    const float a = t[0], bb = t[1];
    t[0] = a * c - bb * s;
    t[1] = a * s + bb * c;
  }
}

// ---------------------------------------------------------------------------
// WMMA fragment helpers (wave32; A/B use this file's contiguous k order, not the ISA table order):
//  A (16x32 f16): lane L -> M = L&15, K = e + 16*(L>>4).
//  B (32x16 f16): lane L -> N = L&15, K = e + 16*(L>>4).
//  C (16x16 f32): lane N holds column N; VGPR j -> M = j + 8*(lane>>4).
// ---------------------------------------------------------------------------
__device__ __forceinline__ v16h load_a_frag(const float* __restrict__ krow,
                                            int c4, int hiOff) {
  const float* rp = krow + c4 * 32 + hiOff;
  const float4 A0 = *(const float4*)(rp + 0);
  const float4 A1 = *(const float4*)(rp + 4);
  const float4 A2 = *(const float4*)(rp + 8);
  const float4 A3 = *(const float4*)(rp + 12);
  v16h a;
  a[0]  = (_Float16)A0.x; a[1]  = (_Float16)A0.y; a[2]  = (_Float16)A0.z; a[3]  = (_Float16)A0.w;
  a[4]  = (_Float16)A1.x; a[5]  = (_Float16)A1.y; a[6]  = (_Float16)A1.z; a[7]  = (_Float16)A1.w;
  a[8]  = (_Float16)A2.x; a[9]  = (_Float16)A2.y; a[10] = (_Float16)A2.z; a[11] = (_Float16)A2.w;
  a[12] = (_Float16)A3.x; a[13] = (_Float16)A3.y; a[14] = (_Float16)A3.z; a[15] = (_Float16)A3.w;
  return a;
}

// ---------------------------------------------------------------------------
// Page scores: one wave per (b, hkv, page).
// score[b][hkv][r][page] = max over the 16 page keys of dot(k, q_{hkv*R+r}).
// A = 16 keys x 128 (4 chained k=32 WMMAs), B columns 0..3 = the R q heads.
// ---------------------------------------------------------------------------
__global__ void page_score_kernel(const float* __restrict__ q,
                                  const float* __restrict__ cache_k,
                                  float* __restrict__ scores) {
  const int gid  = blockIdx.x;
  const int page = gid % NPAGES_;
  const int bh   = gid / NPAGES_;
  const int b    = bh / HKV_;
  const int hkv  = bh % HKV_;
  const int lane = threadIdx.x;
  const int n    = lane & 15;
  const int hi   = lane >> 4;
  const int mkey = lane & 15;
  const int hiOff = hi * 16;

  // B fragments: columns < R carry the roped q heads, rest zero.
  v16h bf[4];
#pragma unroll
  for (int c4 = 0; c4 < 4; ++c4) {
#pragma unroll
    for (int e = 0; e < 16; ++e) {
      const int K = c4 * 32 + hi * 16 + e;
      bf[c4][e] = (n < R_) ? (_Float16)q[((size_t)(b * H_ + hkv * R_ + n)) * D_ + K]
                           : (_Float16)0.0f;
    }
  }

  const int tok = page * PAGE_ + mkey;
  const float* krow = cache_k + (((size_t)b * START_ + tok) * HKV_ + hkv) * D_;

  v8f c = {};
#pragma unroll
  for (int c4 = 0; c4 < 4; ++c4) {
    const v16h a = load_a_frag(krow, c4, hiOff);
    c = __builtin_amdgcn_wmma_f32_16x16x32_f16(false, a, false, bf[c4],
                                               (short)0, c, false, false);
  }

  // per-column max over 16 keys: local over 8 rows, then fold lane^16.
  float lm = c[0];
#pragma unroll
  for (int j = 1; j < 8; ++j) lm = fmaxf(lm, c[j]);
  lm = fmaxf(lm, __shfl_xor(lm, 16, 32));
  if (lane < R_)
    scores[(((size_t)(b * HKV_ + hkv)) * R_ + lane) * NPAGES_ + page] = lm;
}

// ---------------------------------------------------------------------------
// Top-256 pages per (b,hkv,r): in-LDS bitonic sort of 2048 (score, page) pairs.
// ---------------------------------------------------------------------------
__global__ void topk_kernel(const float* __restrict__ scores,
                            int* __restrict__ top) {
  __shared__ float s[2048];
  __shared__ int   idx[2048];
  const int g = blockIdx.x; // (b*HKV+hkv)*R + r
  const float* sc = scores + (size_t)g * NPAGES_;
  for (int i = threadIdx.x; i < 2048; i += 256) {
    s[i]   = (i < NPAGES_) ? sc[i] : -INFINITY;
    idx[i] = i;
  }
  __syncthreads();
  for (int k = 2; k <= 2048; k <<= 1) {
    for (int j = k >> 1; j > 0; j >>= 1) {
      for (int t = threadIdx.x; t < 2048; t += 256) {
        const int ixj = t ^ j;
        if (ixj > t) {
          const bool up = ((t & k) == 0); // descending sort
          const float a = s[t], bb = s[ixj];
          if (up ? (a < bb) : (a > bb)) {
            s[t] = bb; s[ixj] = a;
            const int ti = idx[t]; idx[t] = idx[ixj]; idx[ixj] = ti;
          }
        }
      }
      __syncthreads();
    }
  }
  for (int i = threadIdx.x; i < TOPT_; i += 256) top[(size_t)g * TOPT_ + i] = idx[i];
}

// ---------------------------------------------------------------------------
// Flash-attention partials: one wave per (b*H+h, chunk).
// chunks 0..7  : 32 top pages each (512 gathered prefix keys)
// chunks 8..15 : 512 contiguous suffix keys each
// chunk 16     : the single new token (k_new/v_new)
// Scores via 4 chained WMMAs per 16-key tile; online softmax; float4 V acc.
// Stores per-chunk normalized output + lse; exact merge done later.
// ---------------------------------------------------------------------------
__global__ void attn_partial_kernel(const float* __restrict__ q,
                                    const float* __restrict__ cache_k,
                                    const float* __restrict__ cache_v,
                                    const float* __restrict__ k_new,
                                    const float* __restrict__ v_new,
                                    const int* __restrict__ top,
                                    float* __restrict__ p_out,
                                    float* __restrict__ p_lse) {
  const int bh    = blockIdx.x;
  const int chunk = blockIdx.y;
  const int b   = bh / H_;
  const int h   = bh % H_;
  const int hkv = h / R_;
  const int r   = h % R_;
  const int lane = threadIdx.x;
  const int n    = lane & 15;
  const int hi   = lane >> 4;
  const int mkey = lane & 15;
  const int hiOff = hi * 16;

  const float* qh = q + (size_t)(b * H_ + h) * D_;
  float* po = p_out + ((size_t)bh * NCHUNK_ + chunk) * D_;

  if (chunk == NCHUNK_ - 1) { // single new-token key: softmax over 1 key
    const float* kn = k_new + (size_t)(b * HKV_ + hkv) * D_;
    const float* vn = v_new + (size_t)(b * HKV_ + hkv) * D_;
    const float4 kq = *(const float4*)(kn + lane * 4);
    const float4 qq = *(const float4*)(qh + lane * 4);
    float p = kq.x * qq.x + kq.y * qq.y + kq.z * qq.z + kq.w * qq.w;
#pragma unroll
    for (int off = 16; off > 0; off >>= 1) p += __shfl_xor(p, off, 32);
    const float4 vv = *(const float4*)(vn + lane * 4);
    po[lane * 4 + 0] = vv.x; po[lane * 4 + 1] = vv.y;
    po[lane * 4 + 2] = vv.z; po[lane * 4 + 3] = vv.w;
    if (lane == 0) p_lse[(size_t)bh * NCHUNK_ + chunk] = p * SCALE_;
    return;
  }

  // B fragments: column 0 = this head's roped q; other columns zero.
  v16h bf[4];
#pragma unroll
  for (int c4 = 0; c4 < 4; ++c4) {
#pragma unroll
    for (int e = 0; e < 16; ++e) {
      const int K = c4 * 32 + hi * 16 + e;
      bf[c4][e] = (n == 0) ? (_Float16)qh[K] : (_Float16)0.0f;
    }
  }

  float m = -INFINITY, l = 0.f;
  float acc0 = 0.f, acc1 = 0.f, acc2 = 0.f, acc3 = 0.f; // dims lane*4 .. +3

  const int* tlist = top + (((size_t)(b * HKV_ + hkv)) * R_ + r) * TOPT_;

  for (int t = 0; t < 32; ++t) {
    int tokBase;
    if (chunk < 8) tokBase = tlist[chunk * 32 + t] * PAGE_;
    else           tokBase = PREFIX_ + (chunk - 8) * 512 + t * 16;

    const float* krow =
        cache_k + (((size_t)b * START_ + tokBase + mkey) * HKV_ + hkv) * D_;

    v8f c = {};
#pragma unroll
    for (int c4 = 0; c4 < 4; ++c4) {
      const v16h a = load_a_frag(krow, c4, hiOff);
      c = __builtin_amdgcn_wmma_f32_16x16x32_f16(false, a, false, bf[c4],
                                                 (short)0, c, false, false);
    }

    // broadcast the 16 scores of column 0 (held by lanes 0 and 16).
    float sc[16];
#pragma unroll
    for (int j = 0; j < 8; ++j) {
      sc[j]     = __shfl(c[j], 0, 32)  * SCALE_;
      sc[8 + j] = __shfl(c[j], 16, 32) * SCALE_;
    }
    float tmax = sc[0];
#pragma unroll
    for (int j = 1; j < 16; ++j) tmax = fmaxf(tmax, sc[j]);

    const float mnew = fmaxf(m, tmax);
    const float corr = __expf(m - mnew);
    l *= corr; acc0 *= corr; acc1 *= corr; acc2 *= corr; acc3 *= corr;

#pragma unroll
    for (int j = 0; j < 16; ++j) {
      const float w = __expf(sc[j] - mnew);
      l += w;
      const float4 vv = *(const float4*)(
          cache_v + (((size_t)b * START_ + tokBase + j) * HKV_ + hkv) * D_ +
          lane * 4);
      acc0 += w * vv.x; acc1 += w * vv.y; acc2 += w * vv.z; acc3 += w * vv.w;
    }
    m = mnew;
  }

  const float inv = 1.0f / l;
  po[lane * 4 + 0] = acc0 * inv; po[lane * 4 + 1] = acc1 * inv;
  po[lane * 4 + 2] = acc2 * inv; po[lane * 4 + 3] = acc3 * inv;
  if (lane == 0) p_lse[(size_t)bh * NCHUNK_ + chunk] = m + __logf(l);
}

// ---------------------------------------------------------------------------
// Merge chunk partials by logsumexp (exact softmax merge).
// ---------------------------------------------------------------------------
__global__ void merge_kernel(const float* __restrict__ p_out,
                             const float* __restrict__ p_lse,
                             float* __restrict__ attn_out) {
  const int bh = blockIdx.x;
  const int d  = threadIdx.x; // 128 threads
  float M = -INFINITY;
  for (int c = 0; c < NCHUNK_; ++c)
    M = fmaxf(M, p_lse[(size_t)bh * NCHUNK_ + c]);
  float den = 0.f, acc = 0.f;
  for (int c = 0; c < NCHUNK_; ++c) {
    const float w = __expf(p_lse[(size_t)bh * NCHUNK_ + c] - M);
    den += w;
    acc += w * p_out[((size_t)bh * NCHUNK_ + c) * D_ + d];
  }
  attn_out[(size_t)bh * D_ + d] = acc / den;
}

// ---------------------------------------------------------------------------
extern "C" void kernel_launch(void* const* d_in, const int* in_sizes, int n_in,
                              void* d_out, int out_size, void* d_ws, size_t ws_size,
                              hipStream_t stream) {
  (void)in_sizes; (void)n_in; (void)out_size; (void)ws_size;
  const float* x        = (const float*)d_in[0];
  const float* cosv     = (const float*)d_in[1];
  const float* sinv     = (const float*)d_in[2];
  const float* cache_k  = (const float*)d_in[3];
  const float* cache_v  = (const float*)d_in[4];
  const float* wq       = (const float*)d_in[5];
  const float* wk       = (const float*)d_in[6];
  const float* wv       = (const float*)d_in[7];
  const float* wvo      = (const float*)d_in[8];
  float* out            = (float*)d_out;

  float* ws    = (float*)d_ws;
  float* q     = ws + OFF_Q;
  float* k_new = ws + OFF_KN;
  float* v_new = ws + OFF_VN;
  float* sc    = ws + OFF_SC;
  float* ao    = ws + OFF_AO;
  float* po    = ws + OFF_PO;
  float* pl    = ws + OFF_PL;
  int*   top   = (int*)(ws + OFF_TOP);

  // 1) projections (GEMV, block per output row)
  gemv_kernel<<<B_ * DIM_, 256, 0, stream>>>(x, wq, q, DIM_);
  gemv_kernel<<<B_ * HKV_ * D_, 256, 0, stream>>>(x, wk, k_new, HKV_ * D_);
  gemv_kernel<<<B_ * HKV_ * D_, 256, 0, stream>>>(x, wv, v_new, HKV_ * D_);

  // 2) RoPE on q and new k
  {
    const int total = B_ * H_ * (D_ / 2) + B_ * HKV_ * (D_ / 2);
    rope_kernel<<<(total + 255) / 256, 256, 0, stream>>>(q, k_new, cosv, sinv);
  }

  // 3) page scores (WMMA), 1 wave per (b,hkv,page)
  page_score_kernel<<<B_ * HKV_ * NPAGES_, 32, 0, stream>>>(q, cache_k, sc);

  // 4) top-256 pages per (b,hkv,r)
  topk_kernel<<<B_ * HKV_ * R_, 256, 0, stream>>>(sc, top);

  // 5) flash-attention partials over 17 chunks per head (WMMA scores)
  attn_partial_kernel<<<dim3(B_ * H_, NCHUNK_), 32, 0, stream>>>(
      q, cache_k, cache_v, k_new, v_new, top, po, pl);

  // 6) LSE merge
  merge_kernel<<<B_ * H_, D_, 0, stream>>>(po, pl, ao);

  // 7) output projection
  gemv_kernel<<<B_ * DIM_, 256, 0, stream>>>(ao, wvo, out, DIM_);
}
